// SparseResidualBlock_24352464569754
// MI455X (gfx1250) — compile-verified, matrix operand order checked by reading, not run on hardware
//
#include <hip/hip_runtime.h>
#include <hip/hip_bf16.h>

typedef __attribute__((ext_vector_type(16))) __bf16 v16bf;
typedef __attribute__((ext_vector_type(8)))  float  v8f;
typedef __attribute__((ext_vector_type(8)))  unsigned short us8;

union Frag {
    us8 u[2];
    v16bf v;
};

// ---------------------------------------------------------------------------
// Repack W[k][cin][cout] (f32, row-major) into WMMA B-fragment order (bf16):
// fragment index f = ((k*2 + kchunk)*4 + ntile); each fragment = 32 lanes x 16 bf16.
// Lane L holds column n = ntile*16 + (L&15); K-values kb..kb+7 then kb+16..kb+23
// with kb = (L<16 ? 0 : 8), within kchunk*32.
// ---------------------------------------------------------------------------
__global__ __launch_bounds__(256)
void repack_w_kernel(const float* __restrict__ W, unsigned short* __restrict__ out, int K)
{
    int t = blockIdx.x * blockDim.x + threadIdx.x;
    int total = K * 2 * 4 * 32;
    if (t >= total) return;
    int lane   = t & 31;
    int frag   = t >> 5;
    int ntile  = frag & 3;
    int kchunk = (frag >> 2) & 1;
    int k      = frag >> 3;

    int n     = ntile * 16 + (lane & 15);
    int kbase = kchunk * 32 + ((lane < 16) ? 0 : 8);

    unsigned short* dst = out + (size_t)frag * 512 + lane * 16;
    const float* Wk = W + (size_t)k * 64 * 64;
#pragma unroll
    for (int j = 0; j < 8; ++j) {
        __bf16 b0 = (__bf16)Wk[(kbase + j) * 64 + n];
        __bf16 b1 = (__bf16)Wk[(kbase + 16 + j) * 64 + n];
        dst[j]     = __builtin_bit_cast(unsigned short, b0);
        dst[j + 8] = __builtin_bit_cast(unsigned short, b1);
    }
}

// ---------------------------------------------------------------------------
// f32 -> bf16 elementwise (for x)
// ---------------------------------------------------------------------------
__global__ __launch_bounds__(256)
void f32_to_bf16_kernel(const float* __restrict__ in, __bf16* __restrict__ out, size_t total)
{
    size_t stride = (size_t)gridDim.x * blockDim.x;
    for (size_t i = (size_t)blockIdx.x * blockDim.x + threadIdx.x; i < total; i += stride)
        out[i] = (__bf16)in[i];
}

// ---------------------------------------------------------------------------
// Sparse conv: one wave (32 lanes) per 16-row tile.
// out[r, :] = sum_k mask[r,k] * hin_bf16[nbr[r,k], :] @ Wk   (C=64)
// A fragments built per ISA 16-bit A 16x32 layout; B from repacked weights.
// ---------------------------------------------------------------------------
__global__ __launch_bounds__(128)
void sparse_conv_wmma_kernel(const __bf16* __restrict__ hin,
                             const int* __restrict__ nbr,
                             const int* __restrict__ maskb,
                             const unsigned short* __restrict__ wrep,
                             float* __restrict__ hout,
                             int nTiles, int K)
{
    int wave = threadIdx.x >> 5;
    int lane = threadIdx.x & 31;
    int tile = blockIdx.x * 4 + wave;
    if (tile >= nTiles) return;   // wave-uniform branch: EXEC stays all-ones

    int r0    = tile * 16;
    int myrow = r0 + (lane & 15);
    int kb    = (lane < 16) ? 0 : 8;

    v8f acc0 = {}, acc1 = {}, acc2 = {}, acc3 = {};
    const unsigned short* hin_us = (const unsigned short*)hin;
    const us8 zero8 = {};

    for (int k = 0; k < K; ++k) {
        int idx = nbr[myrow * K + k];
        int m   = maskb[myrow * K + k];
        const unsigned short* src = hin_us + (size_t)idx * 64;
#pragma unroll
        for (int kc = 0; kc < 2; ++kc) {
            Frag a;
            a.u[0] = *(const us8*)(src + kc * 32 + kb);
            a.u[1] = *(const us8*)(src + kc * 32 + kb + 16);
            if (m == 0) { a.u[0] = zero8; a.u[1] = zero8; }

            const unsigned short* wbase =
                wrep + ((size_t)k * 2 + kc) * 4 * 512 + lane * 16;

            Frag b0f, b1f, b2f, b3f;
            b0f.u[0] = *(const us8*)(wbase + 0 * 512);
            b0f.u[1] = *(const us8*)(wbase + 0 * 512 + 8);
            b1f.u[0] = *(const us8*)(wbase + 1 * 512);
            b1f.u[1] = *(const us8*)(wbase + 1 * 512 + 8);
            b2f.u[0] = *(const us8*)(wbase + 2 * 512);
            b2f.u[1] = *(const us8*)(wbase + 2 * 512 + 8);
            b3f.u[0] = *(const us8*)(wbase + 3 * 512);
            b3f.u[1] = *(const us8*)(wbase + 3 * 512 + 8);

            acc0 = __builtin_amdgcn_wmma_f32_16x16x32_bf16(false, a.v, false, b0f.v, (short)0, acc0, false, false);
            acc1 = __builtin_amdgcn_wmma_f32_16x16x32_bf16(false, a.v, false, b1f.v, (short)0, acc1, false, false);
            acc2 = __builtin_amdgcn_wmma_f32_16x16x32_bf16(false, a.v, false, b2f.v, (short)0, acc2, false, false);
            acc3 = __builtin_amdgcn_wmma_f32_16x16x32_bf16(false, a.v, false, b3f.v, (short)0, acc3, false, false);
        }
    }

    // C/D layout: VGPR j -> row r0 + j + (lane<16 ? 0 : 8), col = ntile*16 + (lane&15)
    int col0 = lane & 15;
    int rowb = r0 + ((lane < 16) ? 0 : 8);
#pragma unroll
    for (int j = 0; j < 8; ++j) {
        float* orow = hout + (size_t)(rowb + j) * 64;
        orow[ 0 + col0] = acc0[j];
        orow[16 + col0] = acc1[j];
        orow[32 + col0] = acc2[j];
        orow[48 + col0] = acc3[j];
    }
}

// ---------------------------------------------------------------------------
// Zero the BN-stats accumulators (256 floats: sums1[128] + sums2[128])
// ---------------------------------------------------------------------------
__global__ void zero_stats_kernel(float* __restrict__ p)
{
    p[threadIdx.x] = 0.0f;
}

// ---------------------------------------------------------------------------
// Per-channel sum / sumsq over N rows of [N][64] f32.
// grid*block is a multiple of 64 so each thread sticks to channel t&63.
// ---------------------------------------------------------------------------
__global__ __launch_bounds__(256)
void bn_stats_kernel(const float* __restrict__ h, float* __restrict__ sums, size_t total)
{
    __shared__ float ssum[256];
    __shared__ float ssq[256];
    size_t stride = (size_t)gridDim.x * 256;
    float s = 0.f, q = 0.f;
    for (size_t i = (size_t)blockIdx.x * 256 + threadIdx.x; i < total; i += stride) {
        float v = h[i];
        s += v;
        q += v * v;
    }
    ssum[threadIdx.x] = s;
    ssq[threadIdx.x]  = q;
    __syncthreads();
    if (threadIdx.x < 64) {
        float a = ssum[threadIdx.x] + ssum[threadIdx.x + 64] + ssum[threadIdx.x + 128] + ssum[threadIdx.x + 192];
        float b = ssq[threadIdx.x]  + ssq[threadIdx.x + 64]  + ssq[threadIdx.x + 128]  + ssq[threadIdx.x + 192];
        atomicAdd(&sums[threadIdx.x], a);
        atomicAdd(&sums[64 + threadIdx.x], b);
    }
}

// ---------------------------------------------------------------------------
// scale = gamma * rsqrt(var + eps); shift = beta - mean*scale
// ---------------------------------------------------------------------------
__global__ void bn_finalize_kernel(const float* __restrict__ sums,
                                   const float* __restrict__ gamma,
                                   const float* __restrict__ beta,
                                   float* __restrict__ ss, float invN)
{
    int c = threadIdx.x;
    if (c < 64) {
        float mean = sums[c] * invN;
        float var  = sums[64 + c] * invN - mean * mean;
        float sc   = gamma[c] * rsqrtf(var + 1e-5f);
        ss[c]      = sc;
        ss[64 + c] = beta[c] - mean * sc;
    }
}

// ---------------------------------------------------------------------------
// h1n = bf16(relu(h*scale + shift))   (feeds conv2's L2-resident gathers)
// ---------------------------------------------------------------------------
__global__ __launch_bounds__(256)
void bn_relu_to_bf16_kernel(const float* __restrict__ h, const float* __restrict__ ss,
                            __bf16* __restrict__ out, size_t total)
{
    size_t stride = (size_t)gridDim.x * blockDim.x;
    for (size_t i = (size_t)blockIdx.x * blockDim.x + threadIdx.x; i < total; i += stride) {
        int c = (int)(i & 63);
        float v = fmaxf(h[i] * ss[c] + ss[64 + c], 0.f);
        out[i] = (__bf16)v;
    }
}

// ---------------------------------------------------------------------------
// out = relu(h*scale + shift + x)   (f32)
// ---------------------------------------------------------------------------
__global__ __launch_bounds__(256)
void bn_add_relu_kernel(const float* __restrict__ h, const float* __restrict__ ss,
                        const float* __restrict__ x, float* __restrict__ out, size_t total)
{
    size_t stride = (size_t)gridDim.x * blockDim.x;
    for (size_t i = (size_t)blockIdx.x * blockDim.x + threadIdx.x; i < total; i += stride) {
        int c = (int)(i & 63);
        float v = h[i] * ss[c] + ss[64 + c] + x[i];
        out[i] = fmaxf(v, 0.f);
    }
}

extern "C" void kernel_launch(void* const* d_in, const int* in_sizes, int n_in,
                              void* d_out, int out_size, void* d_ws, size_t ws_size,
                              hipStream_t stream)
{
    const float* x     = (const float*)d_in[0];
    const int*   nbr   = (const int*)d_in[1];
    const int*   maskb = (const int*)d_in[2];
    const float* W1    = (const float*)d_in[3];
    const float* g1    = (const float*)d_in[4];
    const float* b1    = (const float*)d_in[5];
    const float* W2    = (const float*)d_in[6];
    const float* g2    = (const float*)d_in[7];
    const float* b2    = (const float*)d_in[8];
    float* out = (float*)d_out;

    const int C = 64;
    const int N = in_sizes[0] / C;
    const int K = in_sizes[1] / N;           // 27
    const size_t NC = (size_t)N * C;

    // workspace layout
    char* ws = (char*)d_ws;
    __bf16* xbf  = (__bf16*)ws;                          // NC * 2 bytes
    __bf16* h1n  = (__bf16*)(ws + NC * 2);               // NC * 2 bytes
    float*  hraw = (float*)(ws + NC * 4);                // NC * 4 bytes (reused conv1/conv2)
    unsigned short* w1r = (unsigned short*)(ws + NC * 8);        // K*4096 ushorts
    unsigned short* w2r = w1r + (size_t)K * 4096;                // K*4096 ushorts
    float* sums1 = (float*)(w2r + (size_t)K * 4096);             // 128 f32
    float* sums2 = sums1 + 128;                                  // 128 f32
    float* ss1   = sums2 + 128;                                  // 128 f32
    float* ss2   = ss1 + 128;                                    // 128 f32

    const int nTiles = N / 16;
    const dim3 convGrid((nTiles + 3) / 4), convBlk(128);
    const dim3 ewGrid(2048), ewBlk(256);
    const int repackThreads = K * 2 * 4 * 32;
    const dim3 rpGrid((repackThreads + 255) / 256), rpBlk(256);
    const float invN = 1.0f / (float)N;

    // prep
    zero_stats_kernel<<<1, 256, 0, stream>>>(sums1);   // zeroes sums1+sums2 (256 floats)
    f32_to_bf16_kernel<<<ewGrid, ewBlk, 0, stream>>>(x, xbf, NC);
    repack_w_kernel<<<rpGrid, rpBlk, 0, stream>>>(W1, w1r, K);
    repack_w_kernel<<<rpGrid, rpBlk, 0, stream>>>(W2, w2r, K);

    // conv1 -> BN1 -> ReLU (bf16 for conv2 gathers)
    sparse_conv_wmma_kernel<<<convGrid, convBlk, 0, stream>>>(xbf, nbr, maskb, w1r, hraw, nTiles, K);
    bn_stats_kernel<<<1024, 256, 0, stream>>>(hraw, sums1, NC);
    bn_finalize_kernel<<<1, 64, 0, stream>>>(sums1, g1, b1, ss1, invN);
    bn_relu_to_bf16_kernel<<<ewGrid, ewBlk, 0, stream>>>(hraw, ss1, h1n, NC);

    // conv2 -> BN2 + identity + ReLU
    sparse_conv_wmma_kernel<<<convGrid, convBlk, 0, stream>>>(h1n, nbr, maskb, w2r, hraw, nTiles, K);
    bn_stats_kernel<<<1024, 256, 0, stream>>>(hraw, sums2, NC);
    bn_finalize_kernel<<<1, 64, 0, stream>>>(sums2, g2, b2, ss2, invN);
    bn_add_relu_kernel<<<ewGrid, ewBlk, 0, stream>>>(hraw, ss2, x, out, NC);
}
